// AnemoneMLP_85718957293662
// MI455X (gfx1250) — 8-bit matrix operand order repaired and checked by reading, compiled, not run on hardware
//
#include <hip/hip_runtime.h>
#include <hip/hip_bf16.h>
#include <stdint.h>


typedef int v8i __attribute__((ext_vector_type(8)));
typedef unsigned int tdm_u32x4 __attribute__((ext_vector_type(4)));
typedef int tdm_i32x8 __attribute__((ext_vector_type(8)));
typedef int tdm_i32x4 __attribute__((ext_vector_type(4)));

#define ACT_EPS 1e-5f
#define RMS_EPSF 1e-6f
#define KSTEP 64
#define LDS_PITCH 80

static __device__ __forceinline__ void tdm_load_2d(
    uint32_t lds_addr, const int8_t* gptr,
    uint32_t tile_w, uint32_t tile_h,
    uint32_t tensor_w, uint32_t tensor_h) {
  const unsigned long long ga = (unsigned long long)(uintptr_t)gptr;
  tdm_u32x4 g0;
  g0[0] = 1u;
  g0[1] = lds_addr;
  g0[2] = (uint32_t)ga;
  g0[3] = (uint32_t)((ga >> 32) & 0x01FFFFFFu) | (2u << 30);
  tdm_i32x8 g1;
  g1[0] = (int)((1u << 20) | (3u << 22) | (3u << 25));
  g1[1] = (int)((tensor_w & 0xFFFFu) << 16);
  g1[2] = (int)((tensor_w >> 16) | ((tensor_h & 0xFFFFu) << 16));
  g1[3] = (int)((tensor_h >> 16) | (tile_w << 16));
  g1[4] = (int)tile_h;
  g1[5] = (int)tensor_w;
  g1[6] = 0;
  g1[7] = 0;
  const tdm_i32x4 z4 = {0, 0, 0, 0};
  const tdm_i32x8 z8 = {0, 0, 0, 0, 0, 0, 0, 0};
  __builtin_amdgcn_tensor_load_to_lds(g0, g1, z4, z4, z8, 0);
}

static __device__ __forceinline__ uint32_t lds_addr_of(const void* p) {
  return (uint32_t)(uintptr_t)p;
}

__global__ __launch_bounds__(256) void k_absmean(const float* __restrict__ w,
                                                 long long n,
                                                 float* __restrict__ partial) {
  __shared__ float red[256];
  float s = 0.f;
  long long stride = (long long)gridDim.x * blockDim.x;
  for (long long i = (long long)blockIdx.x * blockDim.x + threadIdx.x; i < n; i += stride)
    s += fabsf(w[i]);
  red[threadIdx.x] = s;
  __syncthreads();
  for (int off = 128; off > 0; off >>= 1) {
    if ((int)threadIdx.x < off) red[threadIdx.x] += red[threadIdx.x + off];
    __syncthreads();
  }
  if (threadIdx.x == 0) partial[blockIdx.x] = red[0];
}

__global__ __launch_bounds__(256) void k_sum_partials(const float* __restrict__ partial,
                                                      int n, float* __restrict__ out) {
  __shared__ float red[256];
  float s = 0.f;
  for (int i = threadIdx.x; i < n; i += 256) s += partial[i];
  red[threadIdx.x] = s;
  __syncthreads();
  for (int off = 128; off > 0; off >>= 1) {
    if ((int)threadIdx.x < off) red[threadIdx.x] += red[threadIdx.x + off];
    __syncthreads();
  }
  if (threadIdx.x == 0) *out = red[0];
}

__global__ __launch_bounds__(256) void k_wquant(const float* __restrict__ w,
                                                int8_t* __restrict__ q,
                                                const float* __restrict__ sum,
                                                float inv_n, long long n4) {
  const float mean = fmaxf(*sum * inv_n, ACT_EPS);
  const float scale = 1.f / mean;
  int* __restrict__ q32 = (int*)q;
  long long stride = (long long)gridDim.x * blockDim.x;
  for (long long i = (long long)blockIdx.x * blockDim.x + threadIdx.x; i < n4; i += stride) {
    const float4 v = *(const float4*)(w + i * 4);
    int q0 = __float2int_rn(fminf(fmaxf(v.x * scale, -1.f), 1.f));
    int q1 = __float2int_rn(fminf(fmaxf(v.y * scale, -1.f), 1.f));
    int q2 = __float2int_rn(fminf(fmaxf(v.z * scale, -1.f), 1.f));
    int q3 = __float2int_rn(fminf(fmaxf(v.w * scale, -1.f), 1.f));
    q32[i] = (q0 & 0xff) | ((q1 & 0xff) << 8) | ((q2 & 0xff) << 16) | ((q3 & 0xff) << 24);
  }
}

__global__ __launch_bounds__(256) void k_act_rowquant(const float* __restrict__ x,
                                                      int8_t* __restrict__ q,
                                                      float* __restrict__ dq, int K) {
  __shared__ float rss[256];
  __shared__ float ram[256];
  const float* __restrict__ row = x + (size_t)blockIdx.x * K;
  float ss = 0.f, am = 0.f;
  for (int i = threadIdx.x; i < K; i += 256) {
    float v = row[i];
    ss += v * v;
    am = fmaxf(am, fabsf(v));
  }
  rss[threadIdx.x] = ss;
  ram[threadIdx.x] = am;
  __syncthreads();
  for (int off = 128; off > 0; off >>= 1) {
    if ((int)threadIdx.x < off) {
      rss[threadIdx.x] += rss[threadIdx.x + off];
      ram[threadIdx.x] = fmaxf(ram[threadIdx.x], ram[threadIdx.x + off]);
    }
    __syncthreads();
  }
  const float rinv = rsqrtf(rss[0] / (float)K + RMS_EPSF);
  const float amax = fmaxf(ram[0] * rinv, ACT_EPS);
  const float scale = 127.f / amax;
  if (threadIdx.x == 0) dq[blockIdx.x] = amax * (1.f / 127.f);
  int* __restrict__ qrow = (int*)(q + (size_t)blockIdx.x * K);
  const float m = rinv * scale;
  for (int i = threadIdx.x * 4; i < K; i += 1024) {
    int p = 0;
#pragma unroll
    for (int j = 0; j < 4; ++j) {
      float v = fminf(fmaxf(row[i + j] * m, -128.f), 127.f);
      p |= (__float2int_rn(v) & 0xff) << (8 * j);
    }
    qrow[i >> 2] = p;
  }
}

static __device__ __forceinline__ v8i load_frag_a(const int8_t* p) {
  int2 a0 = *(const int2*)(p);
  int2 a1 = *(const int2*)(p + 8);
  int2 a2 = *(const int2*)(p + 32);
  int2 a3 = *(const int2*)(p + 40);
  v8i a;
  a[0] = a0.x; a[1] = a0.y; a[2] = a1.x; a[3] = a1.y;
  a[4] = a2.x; a[5] = a2.y; a[6] = a3.x; a[7] = a3.y;
  return a;
}
static __device__ __forceinline__ v8i load_frag_b(const int8_t* p) {
  int4 b0 = *(const int4*)(p);
  int4 b1 = *(const int4*)(p + 32);
  v8i b;
  b[0] = b0.x; b[1] = b0.y; b[2] = b0.z; b[3] = b0.w;
  b[4] = b1.x; b[5] = b1.y; b[6] = b1.z; b[7] = b1.w;
  return b;
}

__global__ __launch_bounds__(256) void k_gemm_gate_up(
    const int8_t* __restrict__ xq, const int8_t* __restrict__ wgq,
    const int8_t* __restrict__ wuq, const float* __restrict__ adq,
    const float* __restrict__ sum_g, const float* __restrict__ sum_u,
    float inv_nw, float* __restrict__ h, int M, int N, int K) {
  const uint32_t A_OFF = 0;
  const uint32_t BG_OFF = 128 * LDS_PITCH;
  const uint32_t BU_OFF = BG_OFF + 64 * LDS_PITCH;
  __shared__ __align__(16) int8_t smem[2][20480];

  const int tid = threadIdx.x;
  const int w = tid >> 5, lane = tid & 31;
  const int half = lane >> 4, l16 = lane & 15;
  const int mq = w & 1, nq = w >> 1;
  const int Mb = M >> 7;
  const int bm = blockIdx.x % Mb, bn = blockIdx.x / Mb;
  const int m_base = bm << 7, n_base = bn << 6;
  const bool issuer = (w == 0);
  const int steps = K / KSTEP;

  if (issuer) {
    tdm_load_2d(lds_addr_of(&smem[0][A_OFF]), xq + (size_t)m_base * K, KSTEP, 128, K, M);
    tdm_load_2d(lds_addr_of(&smem[0][BG_OFF]), wgq + (size_t)n_base * K, KSTEP, 64, K, N);
    tdm_load_2d(lds_addr_of(&smem[0][BU_OFF]), wuq + (size_t)n_base * K, KSTEP, 64, K, N);
  }

  v8i accg[4], accu[4];
#pragma unroll
  for (int t = 0; t < 4; ++t) {
    accg[t] = (v8i){0, 0, 0, 0, 0, 0, 0, 0};
    accu[t] = (v8i){0, 0, 0, 0, 0, 0, 0, 0};
  }

  for (int s = 0; s < steps; ++s) {
    if (issuer) __builtin_amdgcn_s_wait_tensorcnt(0);
    __syncthreads();
    if (issuer && (s + 1 < steps)) {
      const int kb2 = (s + 1) * KSTEP;
      int8_t* nb = smem[(s + 1) & 1];
      tdm_load_2d(lds_addr_of(nb + A_OFF), xq + (size_t)m_base * K + kb2, KSTEP, 128, K, M);
      tdm_load_2d(lds_addr_of(nb + BG_OFF), wgq + (size_t)n_base * K + kb2, KSTEP, 64, K, N);
      tdm_load_2d(lds_addr_of(nb + BU_OFF), wuq + (size_t)n_base * K + kb2, KSTEP, 64, K, N);
    }
    const int8_t* buf = smem[s & 1];
    const v8i fg = load_frag_b(buf + BG_OFF + (nq * 16 + l16) * LDS_PITCH + (half << 4));
    const v8i fu = load_frag_b(buf + BU_OFF + (nq * 16 + l16) * LDS_PITCH + (half << 4));
#pragma unroll
    for (int tm = 0; tm < 4; ++tm) {
      const v8i a = load_frag_a(buf + A_OFF + (mq * 64 + tm * 16 + l16) * LDS_PITCH + (half << 4));
      accg[tm] = __builtin_amdgcn_wmma_i32_16x16x64_iu8(true, a, true, fg, accg[tm], false, false);
      accu[tm] = __builtin_amdgcn_wmma_i32_16x16x64_iu8(true, a, true, fu, accu[tm], false, false);
    }
  }

  const float dqg = fmaxf(*sum_g * inv_nw, ACT_EPS);
  const float dqu = fmaxf(*sum_u * inv_nw, ACT_EPS);
  const int n = n_base + nq * 16 + l16;
#pragma unroll
  for (int tm = 0; tm < 4; ++tm) {
#pragma unroll
    for (int r = 0; r < 8; ++r) {
      const int m = m_base + mq * 64 + tm * 16 + (half << 3) + r;
      const float ad = adq[m];
      float g = (float)accg[tm][r] * (ad * dqg);
      g = g / (1.f + expf(-g));
      const float u = (float)accu[tm][r] * (ad * dqu);
      h[(size_t)m * N + n] = g * u;
    }
  }
}

__global__ __launch_bounds__(256) void k_gemm_down(
    const int8_t* __restrict__ hq, const int8_t* __restrict__ wdq,
    const float* __restrict__ hdq, const float* __restrict__ sum_d,
    float inv_nw, float* __restrict__ out, int M, int N, int K) {
  const uint32_t A_OFF = 0;
  const uint32_t B_OFF = 128 * LDS_PITCH;
  __shared__ __align__(16) int8_t smem[2][15360];

  const int tid = threadIdx.x;
  const int w = tid >> 5, lane = tid & 31;
  const int half = lane >> 4, l16 = lane & 15;
  const int mq = w & 1, nq = w >> 1;
  const int Mb = M >> 7;
  const int bm = blockIdx.x % Mb, bn = blockIdx.x / Mb;
  const int m_base = bm << 7, n_base = bn << 6;
  const bool issuer = (w == 0);
  const int steps = K / KSTEP;

  if (issuer) {
    tdm_load_2d(lds_addr_of(&smem[0][A_OFF]), hq + (size_t)m_base * K, KSTEP, 128, K, M);
    tdm_load_2d(lds_addr_of(&smem[0][B_OFF]), wdq + (size_t)n_base * K, KSTEP, 64, K, N);
  }

  v8i acc[4];
#pragma unroll
  for (int t = 0; t < 4; ++t) acc[t] = (v8i){0, 0, 0, 0, 0, 0, 0, 0};

  for (int s = 0; s < steps; ++s) {
    if (issuer) __builtin_amdgcn_s_wait_tensorcnt(0);
    __syncthreads();
    if (issuer && (s + 1 < steps)) {
      const int kb2 = (s + 1) * KSTEP;
      int8_t* nb = smem[(s + 1) & 1];
      tdm_load_2d(lds_addr_of(nb + A_OFF), hq + (size_t)m_base * K + kb2, KSTEP, 128, K, M);
      tdm_load_2d(lds_addr_of(nb + B_OFF), wdq + (size_t)n_base * K + kb2, KSTEP, 64, K, N);
    }
    const int8_t* buf = smem[s & 1];
    const v8i fb = load_frag_b(buf + B_OFF + (nq * 16 + l16) * LDS_PITCH + (half << 4));
#pragma unroll
    for (int tm = 0; tm < 4; ++tm) {
      const v8i a = load_frag_a(buf + A_OFF + (mq * 64 + tm * 16 + l16) * LDS_PITCH + (half << 4));
      acc[tm] = __builtin_amdgcn_wmma_i32_16x16x64_iu8(true, a, true, fb, acc[tm], false, false);
    }
  }

  const float dqw = fmaxf(*sum_d * inv_nw, ACT_EPS);
  const int n = n_base + nq * 16 + l16;
#pragma unroll
  for (int tm = 0; tm < 4; ++tm) {
#pragma unroll
    for (int r = 0; r < 8; ++r) {
      const int m = m_base + mq * 64 + tm * 16 + (half << 3) + r;
      out[(size_t)m * N + n] = (float)acc[tm][r] * (hdq[m] * dqw);
    }
  }
}

extern "C" void kernel_launch(void* const* d_in, const int* in_sizes, int n_in,
                              void* d_out, int out_size, void* d_ws, size_t ws_size,
                              hipStream_t stream) {
  (void)n_in; (void)out_size; (void)ws_size;
  const int D = 2048, F = 8192;
  const int M = in_sizes[0] / D;
  const long long NW = (long long)F * D;

  const float* x = (const float*)d_in[0];
  const float* w_gate = (const float*)d_in[1];
  const float* w_up = (const float*)d_in[2];
  const float* w_down = (const float*)d_in[3];
  float* out = (float*)d_out;

  uint8_t* ws = (uint8_t*)d_ws;
  size_t o = 0;
  float* scal = (float*)(ws + o);  o += 256;
  float* part = (float*)(ws + o);  o += 3 * 2048 * sizeof(float);
  float* adq = (float*)(ws + o);   o += (size_t)M * sizeof(float);
  float* hdq = (float*)(ws + o);   o += (size_t)M * sizeof(float);
  int8_t* xq = (int8_t*)(ws + o);  o += (size_t)M * D;
  int8_t* wgq = (int8_t*)(ws + o); o += (size_t)NW;
  int8_t* wuq = (int8_t*)(ws + o); o += (size_t)NW;
  int8_t* wdq = (int8_t*)(ws + o); o += (size_t)NW;
  int8_t* hq = (int8_t*)(ws + o);  o += (size_t)M * F;
  float* h = (float*)(ws + o);     o += (size_t)M * F * sizeof(float);

  const float inv_nw = 1.f / (float)NW;

  k_absmean<<<2048, 256, 0, stream>>>(w_gate, NW, part + 0 * 2048);
  k_absmean<<<2048, 256, 0, stream>>>(w_up, NW, part + 1 * 2048);
  k_absmean<<<2048, 256, 0, stream>>>(w_down, NW, part + 2 * 2048);
  k_sum_partials<<<1, 256, 0, stream>>>(part + 0 * 2048, 2048, scal + 0);
  k_sum_partials<<<1, 256, 0, stream>>>(part + 1 * 2048, 2048, scal + 1);
  k_sum_partials<<<1, 256, 0, stream>>>(part + 2 * 2048, 2048, scal + 2);

  k_wquant<<<4096, 256, 0, stream>>>(w_gate, wgq, scal + 0, inv_nw, NW / 4);
  k_wquant<<<4096, 256, 0, stream>>>(w_up, wuq, scal + 1, inv_nw, NW / 4);
  k_wquant<<<4096, 256, 0, stream>>>(w_down, wdq, scal + 2, inv_nw, NW / 4);

  k_act_rowquant<<<M, 256, 0, stream>>>(x, xq, adq, D);

  k_gemm_gate_up<<<(M / 128) * (F / 64), 256, 0, stream>>>(
      xq, wgq, wuq, adq, scal + 0, scal + 1, inv_nw, h, M, F, D);

  k_act_rowquant<<<M, 256, 0, stream>>>(h, hq, hdq, F);

  k_gemm_down<<<(M / 128) * (D / 64), 256, 0, stream>>>(
      hq, wdq, hdq, scal + 2, inv_nw, out, M, D, F);
}
